// MultiHeadAttention_56925496541794
// MI455X (gfx1250) — matrix operand order repaired and checked by reading, not compiled, not run on hardware
//
#include <hip/hip_runtime.h>
#include <hip/hip_bf16.h>
#include <stdint.h>

// Problem constants (match reference)
#define NA   8
#define D    512
#define DA   64
#define BSZ  4
#define THW  1024
#define NTOK (BSZ * THW)      // 4096
#define EPS  1e-5f

typedef __attribute__((ext_vector_type(16))) _Float16 v16h;
typedef __attribute__((ext_vector_type(8)))  _Float16 v8h;
typedef __attribute__((ext_vector_type(8)))  float    v8f;

#define SCHED_FENCE() __builtin_amdgcn_sched_barrier(0)

// ---------------------------------------------------------------------------
// CDNA5 async copy: global -> LDS, 16 bytes per lane, tracked by ASYNCcnt.
// ---------------------------------------------------------------------------
__device__ __forceinline__ void async_copy_b128(void* lds_dst, const void* gsrc) {
  asm volatile("global_load_async_to_lds_b128 %0, %1, off"
               :
               : "v"((uint32_t)(uintptr_t)lds_dst),
                 "v"((unsigned long long)(uintptr_t)gsrc)
               : "memory");
}
#define WAIT_ASYNC(imm) asm volatile("s_wait_asynccnt " #imm ::: "memory")

// ---------------------------------------------------------------------------
// WMMA helper: D = A(16x32 f16) * B(32x16 f16) + C(16x16 f32)
// ---------------------------------------------------------------------------
__device__ __forceinline__ v8f wmma16(v16h a, v16h b, v8f c) {
  return __builtin_amdgcn_wmma_f32_16x16x32_f16(
      false, a, false, b, (short)0, c, false, false);
}

// A-fragment loader (16x32 f16). rowptr = start of row M = lane%16.
__device__ __forceinline__ v16h load_a_frag(const _Float16* rowptr, int kbase) {
  const int lane = threadIdx.x & 31;
  const int off  = kbase + ((lane >> 4) << 3);
  v8h lo = *(const v8h*)(rowptr + off);
  v8h hi = *(const v8h*)(rowptr + off + 16);
  v16h r;
#pragma unroll
  for (int i = 0; i < 8; ++i) { r[i] = lo[i]; r[i + 8] = hi[i]; }
  return r;
}

// B-fragment loader (32x16 f16). colptr = k-contiguous data for column lane%16.
__device__ __forceinline__ v16h load_b_frag(const _Float16* colptr, int kbase) {
  const int lane = threadIdx.x & 31;
  const int off  = kbase + ((lane >> 4) << 3);
  v8h lo = *(const v8h*)(colptr + off);
  v8h hi = *(const v8h*)(colptr + off + 16);
  v16h r;
#pragma unroll
  for (int i = 0; i < 8; ++i) { r[i] = lo[i]; r[i + 8] = hi[i]; }
  return r;
}

// Reductions within each 16-lane half-group.
__device__ __forceinline__ float rowmax16(float v) {
#pragma unroll
  for (int m = 8; m >= 1; m >>= 1) v = fmaxf(v, __shfl_xor(v, m, 32));
  return v;
}
__device__ __forceinline__ float rowsum16(float v) {
#pragma unroll
  for (int m = 8; m >= 1; m >>= 1) v += __shfl_xor(v, m, 32);
  return v;
}

// ---------------------------------------------------------------------------
// 1) LayerNorm: x[4096,512] f32 -> xn f16 (one block per token)
// ---------------------------------------------------------------------------
__global__ void ln_kernel(const float* __restrict__ x,
                          const float* __restrict__ gamma,
                          const float* __restrict__ beta,
                          _Float16* __restrict__ xn) {
  const int row = blockIdx.x;
  const int tid = threadIdx.x;
  const float* xr = x + (size_t)row * D;
  float a0 = xr[tid];
  float a1 = xr[tid + 256];
  __shared__ float s1[256], s2[256];
  s1[tid] = a0 + a1;
  s2[tid] = a0 * a0 + a1 * a1;
  __syncthreads();
#pragma unroll
  for (int s = 128; s > 0; s >>= 1) {
    if (tid < s) { s1[tid] += s1[tid + s]; s2[tid] += s2[tid + s]; }
    __syncthreads();
  }
  const float mu   = s1[0] * (1.0f / (float)D);
  const float var  = s2[0] * (1.0f / (float)D) - mu * mu;
  const float rstd = rsqrtf(var + EPS);
  _Float16* xo = xn + (size_t)row * D;
  xo[tid]       = (_Float16)((a0 - mu) * rstd * gamma[tid] + beta[tid]);
  xo[tid + 256] = (_Float16)((a1 - mu) * rstd * gamma[tid + 256] + beta[tid + 256]);
}

// ---------------------------------------------------------------------------
// 2) Weight prep: w[NA][D][DA] f32 -> wT[NA][DA][D] f16 ; w_proj cast.
// ---------------------------------------------------------------------------
__global__ void wt_transpose_kernel(const float* __restrict__ w,
                                    _Float16* __restrict__ wT) {
  const int idx = blockIdx.x * 256 + threadIdx.x;  // NA*D*DA = 262144
  const int e  = idx & (DA - 1);
  const int dd = (idx >> 6) & (D - 1);
  const int h  = idx >> 15;
  wT[((size_t)h * DA + e) * D + dd] = (_Float16)w[idx];
}
__global__ void cast_h_kernel(const float* __restrict__ w,
                              _Float16* __restrict__ wh, int n) {
  const int idx = blockIdx.x * 256 + threadIdx.x;
  if (idx < n) wh[idx] = (_Float16)w[idx];
}

// ---------------------------------------------------------------------------
// 3) QKV projections with async double-buffered weight staging in LDS.
// grid = NA * (NTOK/128); block = 256 (8 waves x 16 tokens)
// v is stored TRANSPOSED: vT[(hb*DA + e)*THW + thw].
// ---------------------------------------------------------------------------
__global__ void __launch_bounds__(256, 1)
qkv_kernel(const _Float16* __restrict__ xn,
           const _Float16* __restrict__ wqT,
           const _Float16* __restrict__ wkT,
           const _Float16* __restrict__ wvT,
           _Float16* __restrict__ q,
           _Float16* __restrict__ k,
           _Float16* __restrict__ vT) {
  const int blk  = blockIdx.x;
  const int tt   = blk & 31;
  const int h    = blk >> 5;
  const int wave = threadIdx.x >> 5;
  const int lane = threadIdx.x & 31;
  const int lg   = lane >> 4;
  const int ln   = lane & 15;
  const int t0   = tt * 128 + wave * 16;

  __shared__ _Float16 Ws[2][3][64 * 32];   // ping-pong weight slices, 24KB

  const _Float16* wT[3] = { wqT + (size_t)h * DA * D,
                            wkT + (size_t)h * DA * D,
                            wvT + (size_t)h * DA * D };
  const int wrow  = threadIdx.x >> 2;
  const int wcol8 = (threadIdx.x & 3) << 3;

  auto issue_w = [&](int kt, int buf) {
#pragma unroll
    for (int m = 0; m < 3; ++m) {
      async_copy_b128(&Ws[buf][m][wrow * 32 + wcol8],
                      wT[m] + (size_t)wrow * D + kt * 32 + wcol8);
    }
  };

  v8f acc[3][4] = {};
  const _Float16* arow = xn + (size_t)(t0 + ln) * D;

  issue_w(0, 0);
  v16h a_cur = load_a_frag(arow, 0);
  for (int kt = 0; kt < D / 32; ++kt) {
    const int buf = kt & 1;
    WAIT_ASYNC(0x0);
    __syncthreads();
    if (kt < D / 32 - 1) issue_w(kt + 1, buf ^ 1);   // overlaps compute below
    v16h a_next = a_cur;
    if (kt < D / 32 - 1) a_next = load_a_frag(arow, (kt + 1) * 32);

    // 1-deep software pipeline; sched fence stops the loads from sinking
    v16h b0 = load_b_frag(&Ws[buf][0][(0 * 16 + ln) * 32], 0);
#pragma unroll
    for (int i = 0; i < 12; ++i) {
      v16h bn;
      if (i < 11) {
        const int m2 = (i + 1) >> 2, nt2 = (i + 1) & 3;
        bn = load_b_frag(&Ws[buf][m2][(nt2 * 16 + ln) * 32], 0);
      }
      SCHED_FENCE();
      acc[i >> 2][i & 3] = wmma16(a_cur, b0, acc[i >> 2][i & 3]);
      if (i < 11) b0 = bn;
    }
    a_cur = a_next;
  }

  const int bb   = t0 / THW;
  const int thw0 = t0 % THW;
  const size_t hb = (size_t)h * BSZ + bb;
#pragma unroll
  for (int m = 0; m < 3; ++m) {
#pragma unroll
    for (int nt = 0; nt < 4; ++nt) {
#pragma unroll
      for (int r = 0; r < 8; ++r) {
        const _Float16 val = (_Float16)acc[m][nt][r];
        const int t = t0 + r + 8 * lg;
        if (m == 0)
          q[((size_t)h * NTOK + t) * DA + nt * 16 + ln] = val;
        else if (m == 1)
          k[((size_t)h * NTOK + t) * DA + nt * 16 + ln] = val;
        else
          vT[(hb * DA + nt * 16 + ln) * THW + thw0 + r + 8 * lg] = val;
      }
    }
  }
}

// ---------------------------------------------------------------------------
// 4) Flash attention with async double-buffered K / V^T tiles.
// grid = NA*BSZ*(THW/128); block = 256 (8 waves x 16 q-rows)
// ---------------------------------------------------------------------------
__global__ void __launch_bounds__(256, 1)
attn_kernel(const _Float16* __restrict__ q,
            const _Float16* __restrict__ k,
            const _Float16* __restrict__ vT,
            const float* __restrict__ Bias,
            const int* __restrict__ Msk,
            _Float16* __restrict__ concat) {
  const int blk  = blockIdx.x;
  const int qt   = blk & 7;
  const int b    = (blk >> 3) & 3;
  const int h    = blk >> 5;
  const int wave = threadIdx.x >> 5;
  const int lane = threadIdx.x & 31;
  const int lg   = lane >> 4;
  const int ln   = lane & 15;

  __shared__ _Float16 Ks[2][128 * DA];     // [krow][d]
  __shared__ _Float16 Vt[2][DA * 128];     // [n][krow]
  __shared__ _Float16 Pb[8][16 * 128];     // per-wave P tile

  const size_t hb  = (size_t)h * BSZ + b;
  const int qrow_g = qt * 128 + wave * 16;
  const int t0     = b * THW + qrow_g;

  auto issue_tile = [&](int kt, int buf) {
    const _Float16* kbase = k + (hb * THW + kt * 128) * DA;
#pragma unroll
    for (int c = 0; c < 4; ++c) {
      const int chunk = threadIdx.x + c * 256;    // 0..1023
      const int row = chunk >> 3, col8 = (chunk & 7) << 3;
      async_copy_b128(&Ks[buf][row * DA + col8], kbase + row * DA + col8);
    }
    const _Float16* vbase = vT + hb * DA * THW + kt * 128;
#pragma unroll
    for (int c = 0; c < 4; ++c) {
      const int chunk = threadIdx.x + c * 256;
      const int n = chunk >> 4, k8 = (chunk & 15) << 3;
      async_copy_b128(&Vt[buf][n * 128 + k8], vbase + (size_t)n * THW + k8);
    }
  };

  const _Float16* qrow = q + (hb * THW + qrow_g + ln) * DA;
  v16h qa0 = load_a_frag(qrow, 0);
  v16h qa1 = load_a_frag(qrow, 32);

  v8f o[4] = {};
  float m8[8], l8[8];
#pragma unroll
  for (int r = 0; r < 8; ++r) { m8[r] = -1e30f; l8[r] = 0.0f; }

  _Float16* Pw = &Pb[wave][0];

  issue_tile(0, 0);
  for (int kt = 0; kt < THW / 128; ++kt) {
    const int buf = kt & 1;

    // warm the cache for this tile's bias block (no counter traffic)
#pragma unroll
    for (int i = 0; i < 2; ++i) {
      const int idx = threadIdx.x + i * 256;   // 0..511
      const int row = idx >> 2, sub = idx & 3;
      __builtin_prefetch(Bias + (hb * THW + qt * 128 + row) * THW +
                             kt * 128 + sub * 32, 0, 1);
    }

    WAIT_ASYNC(0x0);
    __syncthreads();
    if (kt < THW / 128 - 1) issue_tile(kt + 1, buf ^ 1);  // overlaps compute

    // ---- scores + online softmax (next subtile's K frags load under softmax)
    v16h kb0 = load_b_frag(&Ks[buf][ln * DA], 0);
    v16h kb1 = load_b_frag(&Ks[buf][ln * DA], 32);
#pragma unroll
    for (int j = 0; j < 8; ++j) {
      v16h nb0, nb1;
      if (j < 7) {
        const _Float16* krow = &Ks[buf][((j + 1) * 16 + ln) * DA];
        nb0 = load_b_frag(krow, 0);
        nb1 = load_b_frag(krow, 32);
      }
      SCHED_FENCE();
      v8f s = {};
      s = wmma16(qa0, kb0, s);
      s = wmma16(qa1, kb1, s);
      if (j < 7) { kb0 = nb0; kb1 = nb1; }

      const int kcol = kt * 128 + j * 16 + ln;
      float corr[8];
#pragma unroll
      for (int r = 0; r < 8; ++r) {
        const int qr = qrow_g + r + 8 * lg;
        const float bias = Bias[(hb * THW + qr) * THW + kcol];
        const int   msk  = Msk[(size_t)qr * THW + kcol];
        float sv = msk ? -10000.0f : (s[r] * 0.125f + bias);  // 1/sqrt(64)
        const float tm   = rowmax16(sv);
        const float mnew = fmaxf(m8[r], tm);
        const float c    = __expf(m8[r] - mnew);
        const float p    = __expf(sv - mnew);
        l8[r] = l8[r] * c + rowsum16(p);
        m8[r] = mnew;
        corr[r] = c;
        Pw[(r + 8 * lg) * 128 + j * 16 + ln] = (_Float16)p;
      }
#pragma unroll
      for (int nt = 0; nt < 4; ++nt)
#pragma unroll
        for (int r = 0; r < 8; ++r) o[nt][r] *= corr[r];
    }

    // ---- O += P * V (fragments pipelined one step ahead, fenced)
    v16h pa_c = load_a_frag(Pw + ln * 128, 0);
    v16h vb_c = load_b_frag(&Vt[buf][ln * 128], 0);
#pragma unroll
    for (int i = 0; i < 16; ++i) {
      const int nt = i & 3;
      v16h pa_n = pa_c, vb_n = vb_c;
      if (i < 15) {
        const int kk2 = (i + 1) >> 2, nt2 = (i + 1) & 3;
        vb_n = load_b_frag(&Vt[buf][(nt2 * 16 + ln) * 128], kk2 * 32);
        if (nt2 == 0) pa_n = load_a_frag(Pw + ln * 128, kk2 * 32);
      }
      SCHED_FENCE();
      o[nt] = wmma16(pa_c, vb_c, o[nt]);
      pa_c = pa_n; vb_c = vb_n;
    }
  }

  float inv[8];
#pragma unroll
  for (int r = 0; r < 8; ++r) inv[r] = 1.0f / l8[r];
#pragma unroll
  for (int nt = 0; nt < 4; ++nt) {
#pragma unroll
    for (int r = 0; r < 8; ++r) {
      const int t = t0 + r + 8 * lg;
      concat[(size_t)t * (NA * DA) + h * DA + nt * 16 + ln] =
          (_Float16)(o[nt][r] * inv[r]);
    }
  }
}

// ---------------------------------------------------------------------------
// 5) Output projection + residual with async double-buffered weights.
// grid = (NTOK/128) * (D/128); block = 256
// ---------------------------------------------------------------------------
__global__ void __launch_bounds__(256, 1)
proj_kernel(const _Float16* __restrict__ concat,
            const _Float16* __restrict__ wp,
            const float* __restrict__ x,
            float* __restrict__ out) {
  const int blk  = blockIdx.x;
  const int ot   = blk & 3;
  const int tt   = blk >> 2;
  const int wave = threadIdx.x >> 5;
  const int lane = threadIdx.x & 31;
  const int lg   = lane >> 4;
  const int ln   = lane & 15;
  const int t0   = tt * 128 + wave * 16;
  const int o0   = ot * 128;

  __shared__ _Float16 Wp[2][128 * 32];   // 16KB ping-pong

  auto issue_w = [&](int kt, int buf) {
#pragma unroll
    for (int i = 0; i < 2; ++i) {
      const int c = threadIdx.x + i * 256;   // 0..511
      const int row = c >> 2, col8 = (c & 3) << 3;
      async_copy_b128(&Wp[buf][row * 32 + col8],
                      wp + (size_t)(o0 + row) * D + kt * 32 + col8);
    }
  };

  v8f acc[8] = {};
  const _Float16* arow = concat + (size_t)(t0 + ln) * (NA * DA);

  issue_w(0, 0);
  v16h a_cur = load_a_frag(arow, 0);
  for (int kt = 0; kt < (NA * DA) / 32; ++kt) {
    const int buf = kt & 1;
    WAIT_ASYNC(0x0);
    __syncthreads();
    if (kt < (NA * DA) / 32 - 1) issue_w(kt + 1, buf ^ 1);
    v16h a_next = a_cur;
    if (kt < (NA * DA) / 32 - 1) a_next = load_a_frag(arow, (kt + 1) * 32);

    v16h b0 = load_b_frag(&Wp[buf][(0 * 16 + ln) * 32], 0);
#pragma unroll
    for (int nt = 0; nt < 8; ++nt) {
      v16h bn;
      if (nt < 7) bn = load_b_frag(&Wp[buf][((nt + 1) * 16 + ln) * 32], 0);
      SCHED_FENCE();
      acc[nt] = wmma16(a_cur, b0, acc[nt]);
      if (nt < 7) b0 = bn;
    }
    a_cur = a_next;
  }
#pragma unroll
  for (int nt = 0; nt < 8; ++nt) {
#pragma unroll
    for (int r = 0; r < 8; ++r) {
      const int t = t0 + r + 8 * lg;
      const int oo = o0 + nt * 16 + ln;
      out[(size_t)t * D + oo] = acc[nt][r] + x[(size_t)t * D + oo];
    }
  }
}

// ---------------------------------------------------------------------------
// Host launcher
// ---------------------------------------------------------------------------
extern "C" void kernel_launch(void* const* d_in, const int* in_sizes, int n_in,
                              void* d_out, int out_size, void* d_ws, size_t ws_size,
                              hipStream_t stream) {
  const float* x      = (const float*)d_in[0];
  const float* Bias   = (const float*)d_in[1];
  const int*   Msk    = (const int*)  d_in[2];
  const float* gamma  = (const float*)d_in[3];
  const float* beta   = (const float*)d_in[4];
  const float* w_q    = (const float*)d_in[5];
  const float* w_k    = (const float*)d_in[6];
  const float* w_v    = (const float*)d_in[7];
  const float* w_proj = (const float*)d_in[8];
  float* out = (float*)d_out;

  char* ws = (char*)d_ws;
  const size_t SZ_XN  = (size_t)NTOK * D * 2;
  const size_t SZ_WT  = (size_t)NA * DA * D * 2;
  const size_t SZ_WP  = (size_t)D * NA * DA * 2;
  const size_t SZ_QKV = (size_t)NA * NTOK * DA * 2;
  const size_t SZ_CC  = (size_t)NTOK * NA * DA * 2;

  _Float16* xn  = (_Float16*)(ws);                    size_t off = SZ_XN;
  _Float16* wqT = (_Float16*)(ws + off); off += SZ_WT;
  _Float16* wkT = (_Float16*)(ws + off); off += SZ_WT;
  _Float16* wvT = (_Float16*)(ws + off); off += SZ_WT;
  _Float16* wpH = (_Float16*)(ws + off); off += SZ_WP;
  _Float16* qb  = (_Float16*)(ws + off); off += SZ_QKV;
  _Float16* kb  = (_Float16*)(ws + off); off += SZ_QKV;
  _Float16* vTb = (_Float16*)(ws + off); off += SZ_QKV;
  _Float16* cc  = (_Float16*)(ws + off); off += SZ_CC;
  (void)off; (void)ws_size; (void)in_sizes; (void)n_in; (void)out_size;

  ln_kernel<<<NTOK, 256, 0, stream>>>(x, gamma, beta, xn);

  wt_transpose_kernel<<<(NA * D * DA) / 256, 256, 0, stream>>>(w_q, wqT);
  wt_transpose_kernel<<<(NA * D * DA) / 256, 256, 0, stream>>>(w_k, wkT);
  wt_transpose_kernel<<<(NA * D * DA) / 256, 256, 0, stream>>>(w_v, wvT);
  cast_h_kernel<<<(D * NA * DA) / 256, 256, 0, stream>>>(w_proj, wpH, D * NA * DA);

  qkv_kernel<<<NA * (NTOK / 128), 256, 0, stream>>>(xn, wqT, wkT, wvT, qb, kb, vTb);

  attn_kernel<<<NA * BSZ * (THW / 128), 256, 0, stream>>>(qb, kb, vTb, Bias, Msk, cc);

  proj_kernel<<<(NTOK / 128) * (D / 128), 256, 0, stream>>>(cc, wpH, x, out);
}
